// CCSequenceModel_77610059038885
// MI455X (gfx1250) — matrix operand order repaired and checked by reading, not compiled, not run on hardware
//
#include <hip/hip_runtime.h>
#include <hip/hip_bf16.h>
#include <cstdint>

// ---------------- problem constants ----------------
#define B_   8192
#define L_   512
#define NIN_ 6
#define H_   64
#define T_   180
#define G3H_ 192

#define HSTR 88          // bf16 elements per LDS h row (padded; 176B, 16B aligned)
#define ROWS_PER_WG 64   // batch rows per workgroup

typedef __attribute__((ext_vector_type(16))) __bf16 v16bf;
typedef __attribute__((ext_vector_type(8)))  __bf16 v8bf;
typedef __attribute__((ext_vector_type(8)))  float  v8f;

// ---------------- small helpers ----------------
__device__ __forceinline__ v8f wmma_bf16(v16bf a, v16bf b, v8f c) {
  // D = A(16x32 bf16) * B(32x16 bf16) + C(16x16 f32)
  return __builtin_amdgcn_wmma_f32_16x16x32_bf16(
      /*neg_a=*/false, a, /*neg_b=*/false, b,
      /*c_mod=*/(short)0, c, /*reuse_a=*/false, /*reuse_b=*/false);
}

__device__ __forceinline__ v16bf zero16bf() {
  v16bf z;
#pragma unroll
  for (int i = 0; i < 16; ++i) z[i] = (__bf16)0.0f;
  return z;
}

__device__ __forceinline__ v8f splat8(float x) {
  v8f v;
#pragma unroll
  for (int i = 0; i < 8; ++i) v[i] = x;
  return v;
}

// fast gate math: single v_exp_f32 + single v_rcp_f32, no IEEE div chains
__device__ __forceinline__ float fsigmoid(float x) {
  return __builtin_amdgcn_rcpf(1.0f + __expf(-x));
}
__device__ __forceinline__ float ftanh(float x) {
  float xc = fminf(15.0f, fmaxf(-15.0f, x));
  float e  = __expf(2.0f * xc);
  return fmaf(-2.0f, __builtin_amdgcn_rcpf(e + 1.0f), 1.0f);
}

// Load one 32x16 bf16 B-operand tile from an fp32 weight matrix W (layout
// (192, ldw) row-major, as in PyTorch W_ih/W_hh). B[k,n] = W[n,k] (we need W^T).
__device__ __forceinline__ v16bf load_btile(const float* __restrict__ W, int ldw,
                                            int n0, int k0, int lane) {
  int col = n0 + (lane & 15);
  int kb  = k0 + ((lane & 16) ? 8 : 0);
  v16bf b;
#pragma unroll
  for (int e = 0; e < 16; ++e) {
    int k = kb + e + ((e >= 8) ? 8 : 0);
    float v = (k < ldw) ? W[col * ldw + k] : 0.0f;
    b[e] = (__bf16)v;
  }
  return b;
}

// Load gate-sliced weight set for one GRU layer into registers (this wave's
// 16-column slice of each gate block).
__device__ __forceinline__ void load_wset(const float* __restrict__ Wih, int ldin,
                                          const float* __restrict__ Whh,
                                          v16bf (&Bih)[3][2], v16bf (&Bhh)[3][2],
                                          int wn, int lane) {
#pragma unroll
  for (int g = 0; g < 3; ++g) {
    int n0 = g * 64 + wn * 16;
#pragma unroll
    for (int kt = 0; kt < 2; ++kt) {
      Bih[g][kt] = load_btile(Wih, ldin, n0, kt * 32, lane);
      Bhh[g][kt] = load_btile(Whh, H_,   n0, kt * 32, lane);
    }
  }
}

// Load one 16x32 bf16 A-operand tile (rows row0..row0+15, K = k0..k0+31)
// from a row-major bf16 LDS buffer with row stride HSTR.
__device__ __forceinline__ v16bf load_atile(const __bf16* hb, int row0, int k0, int lane) {
  int row = row0 + (lane & 15);
  int kb  = k0 + ((lane & 16) ? 8 : 0);
  const v8bf* p = (const v8bf*)(hb + row * HSTR + kb);
  v8bf lo = p[0];   // K = kb .. kb+7   -> elements 0..7
  v8bf hi = p[2];   // K = kb+16..kb+23 -> elements 8..15
  return __builtin_shufflevector(lo, hi, 0,1,2,3,4,5,6,7,8,9,10,11,12,13,14,15);
}

// Store this wave's 16x16 f32 accumulator-layout tile into the bf16 LDS h buffer.
__device__ __forceinline__ void store_htile(__bf16* hb, int row0, int col, int lane, v8f h) {
  int rbase = row0 + ((lane & 16) ? 8 : 0);
#pragma unroll
  for (int k = 0; k < 8; ++k) hb[(rbase + k) * HSTR + col] = (__bf16)h[k];
}

// Build padded A tile from x_t (K=0..5 valid). rowG0 = global batch row of M=0.
__device__ __forceinline__ v16bf build_ax(const float* __restrict__ X, int rowG0,
                                          int t, int lane) {
  v16bf a = zero16bf();
  if ((lane & 16) == 0) {
    const float* xp = X + ((size_t)(rowG0 + (lane & 15)) * L_ + t) * NIN_;
#pragma unroll
    for (int e = 0; e < NIN_; ++e) a[e] = (__bf16)xp[e];
    if (t + 16 < L_) __builtin_prefetch(xp + 16 * NIN_, 0, 0);
  }
  return a;
}

// Build padded A tile from the scalar decoder feedback cv (K=0 only valid).
__device__ __forceinline__ v16bf build_acv(const float* cvbuf, int rowL0, int lane) {
  v16bf a = zero16bf();
  if ((lane & 16) == 0) a[0] = (__bf16)cvbuf[rowL0 + (lane & 15)];
  return a;
}

// One GRU cell for one 16-row M tile (this wave's 16 hidden columns).
template <int NKIN>
__device__ __forceinline__ void gru_mtile(const v16bf (&ax)[2], const v16bf (&ah)[2],
                                          const v16bf (&Bih)[3][2], const v16bf (&Bhh)[3][2],
                                          float br, float bz, float bi, float bn,
                                          v8f& h) {
  v8f ar = splat8(br);   // ir + hr (+ biases)
  v8f az = splat8(bz);   // iz + hz (+ biases)
  v8f ai = splat8(bi);   // in (input side, + bih_n)
  v8f an = splat8(bn);   // hn (hidden side, + bhh_n)
#pragma unroll
  for (int kt = 0; kt < NKIN; ++kt) {
    ar = wmma_bf16(ax[kt], Bih[0][kt], ar);
    az = wmma_bf16(ax[kt], Bih[1][kt], az);
    ai = wmma_bf16(ax[kt], Bih[2][kt], ai);
  }
#pragma unroll
  for (int kt = 0; kt < 2; ++kt) {
    ar = wmma_bf16(ah[kt], Bhh[0][kt], ar);
    az = wmma_bf16(ah[kt], Bhh[1][kt], az);
    an = wmma_bf16(ah[kt], Bhh[2][kt], an);
  }
#pragma unroll
  for (int e = 0; e < 8; ++e) {
    float r = fsigmoid(ar[e]);
    float z = fsigmoid(az[e]);
    float n = ftanh(ai[e] + r * an[e]);
    h[e] = (1.0f - z) * n + z * h[e];
  }
}

// ---------------- the persistent seq2seq kernel ----------------
__global__ __launch_bounds__(256, 1)
void cc_seq2seq_kernel(
    const float* __restrict__ X,
    const float* __restrict__ e0Wih, const float* __restrict__ e0Whh,
    const float* __restrict__ e0bih, const float* __restrict__ e0bhh,
    const float* __restrict__ e1Wih, const float* __restrict__ e1Whh,
    const float* __restrict__ e1bih, const float* __restrict__ e1bhh,
    const float* __restrict__ d0Wih, const float* __restrict__ d0Whh,
    const float* __restrict__ d0bih, const float* __restrict__ d0bhh,
    const float* __restrict__ d1Wih, const float* __restrict__ d1Whh,
    const float* __restrict__ d1bih, const float* __restrict__ d1bhh,
    const float* __restrict__ onW, const float* __restrict__ onB,
    const float* __restrict__ cvW, const float* __restrict__ cvB,
    float* __restrict__ Out) {

  // ping-pong double buffers: [layer][parity]
  __shared__ __bf16 hbuf[2][2][ROWS_PER_WG * HSTR];
  __shared__ float  cvbuf[ROWS_PER_WG];
  __shared__ float  wonb[H_];
  __shared__ float  wcvb[H_];

  const int tid  = threadIdx.x;
  const int lane = tid & 31;
  const int wave = tid >> 5;
  const int wn   = wave & 3;   // hidden-column group (16 cols)
  const int wm   = wave >> 2;  // batch-row group (32 rows)
  const int wrow = wm * 32;    // local row base of this wave
  const int colh = wn * 16 + (lane & 15);  // this lane's hidden column
  const int blockRow0 = blockIdx.x * ROWS_PER_WG;

  // ---- init LDS state ----
  for (int i = tid; i < ROWS_PER_WG * HSTR; i += 256) {
    hbuf[0][0][i] = (__bf16)0.0f;
    hbuf[0][1][i] = (__bf16)0.0f;
    hbuf[1][0][i] = (__bf16)0.0f;
    hbuf[1][1][i] = (__bf16)0.0f;
  }
  if (tid < ROWS_PER_WG) cvbuf[tid] = 0.0f;
  if (tid < H_) { wonb[tid] = onW[tid]; wcvb[tid] = cvW[tid]; }
  const float onBv = onB[0];
  const float cvBv = cvB[0];
  __syncthreads();

  // ---- encoder weights -> registers (bf16 B-operand tiles) ----
  v16bf Wih0[3][2], Whh0[3][2], Wih1[3][2], Whh1[3][2];
  load_wset(e0Wih, NIN_, e0Whh, Wih0, Whh0, wn, lane);
  load_wset(e1Wih, H_,   e1Whh, Wih1, Whh1, wn, lane);

  // per-lane gate biases (column = colh)
  const int cg = colh;
  float b0r = e0bih[cg] + e0bhh[cg];
  float b0z = e0bih[64 + cg] + e0bhh[64 + cg];
  float b0i = e0bih[128 + cg];
  float b0n = e0bhh[128 + cg];
  float b1r = e1bih[cg] + e1bhh[cg];
  float b1z = e1bih[64 + cg] + e1bhh[64 + cg];
  float b1i = e1bih[128 + cg];
  float b1n = e1bhh[128 + cg];

  // fp32 hidden state, accumulator layout, this wave's 16 columns x 2 M tiles
  v8f h0reg[2], h1reg[2];
#pragma unroll
  for (int mt = 0; mt < 2; ++mt) { h0reg[mt] = splat8(0.0f); h1reg[mt] = splat8(0.0f); }

  // ================= encoder: 1 barrier per timestep (ping-pong) =============
  for (int t = 0; t < L_; ++t) {
    const int cur = t & 1, nxt = cur ^ 1;
    const __bf16* h0r = hbuf[0][cur];
    __bf16*       h0w = hbuf[0][nxt];
    const __bf16* h1r = hbuf[1][cur];
    __bf16*       h1w = hbuf[1][nxt];

    // ---- layer 0: input = x_t, hidden = h0r -> h0w ----
    v16bf ax[2][2], ah[2][2];
#pragma unroll
    for (int mt = 0; mt < 2; ++mt) {
      ax[mt][0] = build_ax(X, blockRow0 + wrow + mt * 16, t, lane);
#pragma unroll
      for (int kt = 0; kt < 2; ++kt)
        ah[mt][kt] = load_atile(h0r, wrow + mt * 16, kt * 32, lane);
    }
#pragma unroll
    for (int mt = 0; mt < 2; ++mt) {
      gru_mtile<1>(ax[mt], ah[mt], Wih0, Whh0, b0r, b0z, b0i, b0n, h0reg[mt]);
      store_htile(h0w, wrow + mt * 16, colh, lane, h0reg[mt]);
    }
    __syncthreads();   // h0w visible; also orders last step's reads vs our writes

    // ---- layer 1: input = h0w (new), hidden = h1r -> h1w ----
    v16bf a0[2][2], a1[2][2];
#pragma unroll
    for (int mt = 0; mt < 2; ++mt)
#pragma unroll
      for (int kt = 0; kt < 2; ++kt) {
        a0[mt][kt] = load_atile(h0w, wrow + mt * 16, kt * 32, lane);
        a1[mt][kt] = load_atile(h1r, wrow + mt * 16, kt * 32, lane);
      }
#pragma unroll
    for (int mt = 0; mt < 2; ++mt) {
      gru_mtile<2>(a0[mt], a1[mt], Wih1, Whh1, b1r, b1z, b1i, b1n, h1reg[mt]);
      store_htile(h1w, wrow + mt * 16, colh, lane, h1reg[mt]);
    }
    // no barrier needed here: next step's barrier orders h1w/h0w hazards
  }
  __syncthreads();   // make encoder-final state visible before decoder

  // ---- swap to decoder weights / biases ----
  load_wset(d0Wih, 1,  d0Whh, Wih0, Whh0, wn, lane);
  load_wset(d1Wih, H_, d1Whh, Wih1, Whh1, wn, lane);
  b0r = d0bih[cg] + d0bhh[cg];
  b0z = d0bih[64 + cg] + d0bhh[64 + cg];
  b0i = d0bih[128 + cg];
  b0n = d0bhh[128 + cg];
  b1r = d1bih[cg] + d1bhh[cg];
  b1z = d1bih[64 + cg] + d1bhh[64 + cg];
  b1i = d1bih[128 + cg];
  b1n = d1bhh[128 + cg];

  // ================= decoder (autoregressive) =================
  // L_ is even, so parity continues seamlessly: state sits in buffer [.][0].
  for (int t = 0; t < T_; ++t) {
    const int cur = t & 1, nxt = cur ^ 1;
    const __bf16* h0r = hbuf[0][cur];
    __bf16*       h0w = hbuf[0][nxt];
    const __bf16* h1r = hbuf[1][cur];
    __bf16*       h1w = hbuf[1][nxt];

    // ---- layer 0: input = previous cv (scalar per row) ----
    v16bf ax[2][2], ah[2][2];
#pragma unroll
    for (int mt = 0; mt < 2; ++mt) {
      ax[mt][0] = build_acv(cvbuf, wrow + mt * 16, lane);
#pragma unroll
      for (int kt = 0; kt < 2; ++kt)
        ah[mt][kt] = load_atile(h0r, wrow + mt * 16, kt * 32, lane);
    }
#pragma unroll
    for (int mt = 0; mt < 2; ++mt) {
      gru_mtile<1>(ax[mt], ah[mt], Wih0, Whh0, b0r, b0z, b0i, b0n, h0reg[mt]);
      store_htile(h0w, wrow + mt * 16, colh, lane, h0reg[mt]);
    }
    __syncthreads();

    // ---- layer 1 ----
    v16bf a0[2][2], a1[2][2];
#pragma unroll
    for (int mt = 0; mt < 2; ++mt)
#pragma unroll
      for (int kt = 0; kt < 2; ++kt) {
        a0[mt][kt] = load_atile(h0w, wrow + mt * 16, kt * 32, lane);
        a1[mt][kt] = load_atile(h1r, wrow + mt * 16, kt * 32, lane);
      }
#pragma unroll
    for (int mt = 0; mt < 2; ++mt) {
      gru_mtile<2>(a0[mt], a1[mt], Wih1, Whh1, b1r, b1z, b1i, b1n, h1reg[mt]);
      store_htile(h1w, wrow + mt * 16, colh, lane, h1reg[mt]);
    }
    __syncthreads();   // h1w visible to head threads

    // ---- output heads: one thread per batch row, 16B vector LDS reads ----
    if (tid < ROWS_PER_WG) {
      const v8bf* hp = (const v8bf*)(h1w + tid * HSTR);
      float don = onBv, dcv = cvBv;
#pragma unroll
      for (int cb = 0; cb < 8; ++cb) {
        v8bf hv8 = hp[cb];
#pragma unroll
        for (int e = 0; e < 8; ++e) {
          float hv = (float)hv8[e];
          don = fmaf(hv, wonb[cb * 8 + e], don);
          dcv = fmaf(hv, wcvb[cb * 8 + e], dcv);
        }
      }
      // sigmoid(don) > 0.5  <=>  don > 0
      float cv = (don > 0.0f) ? dcv : 0.0f;
      cvbuf[tid] = cv;
      Out[(size_t)(blockRow0 + tid) * T_ + t] = cv;
    }
    __syncthreads();   // cv feedback visible to all waves
  }
}

// ---------------- host-side launcher ----------------
extern "C" void kernel_launch(void* const* d_in, const int* in_sizes, int n_in,
                              void* d_out, int out_size, void* d_ws, size_t ws_size,
                              hipStream_t stream) {
  (void)in_sizes; (void)n_in; (void)out_size; (void)d_ws; (void)ws_size;
  const float* X     = (const float*)d_in[0];
  const float* e0Wih = (const float*)d_in[1];
  const float* e0Whh = (const float*)d_in[2];
  const float* e0bih = (const float*)d_in[3];
  const float* e0bhh = (const float*)d_in[4];
  const float* e1Wih = (const float*)d_in[5];
  const float* e1Whh = (const float*)d_in[6];
  const float* e1bih = (const float*)d_in[7];
  const float* e1bhh = (const float*)d_in[8];
  const float* d0Wih = (const float*)d_in[9];
  const float* d0Whh = (const float*)d_in[10];
  const float* d0bih = (const float*)d_in[11];
  const float* d0bhh = (const float*)d_in[12];
  const float* d1Wih = (const float*)d_in[13];
  const float* d1Whh = (const float*)d_in[14];
  const float* d1bih = (const float*)d_in[15];
  const float* d1bhh = (const float*)d_in[16];
  const float* onW   = (const float*)d_in[17];
  const float* onB   = (const float*)d_in[18];
  const float* cvW   = (const float*)d_in[19];
  const float* cvB   = (const float*)d_in[20];
  float* Out = (float*)d_out;

  dim3 grid(B_ / ROWS_PER_WG);   // 128 workgroups, each owns 64 batch rows
  dim3 block(256);               // 8 wave32s: 2(M) x 4(N)
  cc_seq2seq_kernel<<<grid, block, 0, stream>>>(
      X,
      e0Wih, e0Whh, e0bih, e0bhh,
      e1Wih, e1Whh, e1bih, e1bhh,
      d0Wih, d0Whh, d0bih, d0bhh,
      d1Wih, d1Whh, d1bih, d1bhh,
      onW, onB, cvW, cvB, Out);
}
